// SioConvPS_13709535609453
// MI455X (gfx1250) — compile-verified, matrix operand order checked by reading, not run on hardware
//
#include <hip/hip_runtime.h>
#include <hip/hip_bf16.h>

#define BB   2
#define LL   2048
#define DIMD 1024
#define FFD  2048
#define VV   32000
#define NDEP 4
#define MTOK (BB * LL)   // 4096 rows

typedef __attribute__((ext_vector_type(16))) __bf16 v16bf;
typedef __attribute__((ext_vector_type(8)))  __bf16 v8bf;
typedef __attribute__((ext_vector_type(8)))  float  v8f;

// ---------------------------------------------------------------- helpers
__device__ __forceinline__ float block_reduce_sum_256(float v) {
    #pragma unroll
    for (int off = 16; off > 0; off >>= 1) v += __shfl_xor(v, off, 32);
    __shared__ float part[8];
    __shared__ float total;
    int w = threadIdx.x >> 5, l = threadIdx.x & 31;
    if (l == 0) part[w] = v;
    __syncthreads();
    if (w == 0) {
        float s = (l < 8) ? part[l] : 0.0f;
        #pragma unroll
        for (int off = 4; off > 0; off >>= 1) s += __shfl_xor(s, off, 32);
        if (l == 0) total = s;
    }
    __syncthreads();
    return total;
}

__device__ __forceinline__ float softplusf(float x) {
    return fmaxf(x, 0.0f) + log1pf(expf(-fabsf(x)));
}
__device__ __forceinline__ float siluf(float x) {
    return x / (1.0f + expf(-x));
}
__device__ __forceinline__ float logaddexpf_(float a, float b) {
    float m = fmaxf(a, b);
    return m + log1pf(expf(fminf(a, b) - m));
}

// 32-bit LDS byte-offset of a pointer into static shared memory (addrspace 3)
__device__ __forceinline__ unsigned lds_addr32(const void* p) {
    return (unsigned)(unsigned long long)
        (const __attribute__((address_space(3))) void*)p;
}

// async DMA: 16 bytes global -> LDS per lane, tracked by ASYNCcnt
__device__ __forceinline__ void async_ld_b128(unsigned lds_off, const __bf16* gaddr) {
    asm volatile("global_load_async_to_lds_b128 %0, %1, off"
                 :: "v"(lds_off), "v"(gaddr) : "memory");
}

// Build a 16-bit WMMA A/B fragment (16x32 MxK layout): lane holds 8 bf16 at
// K = khalf*8 and 8 bf16 at K = 16 + khalf*8 (both contiguous 16B chunks).
__device__ __forceinline__ v16bf load_frag16(const __bf16* p) {
    v8bf lo = *(const v8bf*)p;
    v8bf hi = *(const v8bf*)(p + 16);
    return __builtin_shufflevector(lo, hi,
        0, 1, 2, 3, 4, 5, 6, 7, 8, 9, 10, 11, 12, 13, 14, 15);
}

// ------------------------------------------------- fp32 -> bf16 transpose
// Wt[n*K + k] = bf16(W[k*N + n]);  grid (N/32, K/32), block (32, 8)
__global__ void cvt_transpose_kernel(const float* __restrict__ W,
                                     __bf16* __restrict__ Wt, int K, int N) {
    __shared__ float tile[32][33];
    int nb = blockIdx.x * 32, kb = blockIdx.y * 32;
    int tx = threadIdx.x, ty = threadIdx.y;
    #pragma unroll
    for (int i = 0; i < 32; i += 8)
        tile[ty + i][tx] = W[(size_t)(kb + ty + i) * N + nb + tx];
    __syncthreads();
    #pragma unroll
    for (int i = 0; i < 32; i += 8)
        Wt[(size_t)(nb + ty + i) * K + kb + tx] = (__bf16)tile[tx][ty + i];
}

// ------------------------------------------------- embedding + renorm
__global__ void embed_renorm_kernel(const int* __restrict__ tokens,
                                    const float* __restrict__ emb,
                                    float* __restrict__ x) {
    int row = blockIdx.x;
    int tok = tokens[row];
    const float* e = emb + (size_t)tok * DIMD;
    float v[4], ss = 0.0f;
    #pragma unroll
    for (int i = 0; i < 4; ++i) {
        v[i] = e[threadIdx.x + i * 256];
        ss += v[i] * v[i];
    }
    ss = block_reduce_sum_256(ss);
    float nrm = sqrtf(ss);
    float scale = (nrm > 1.0f) ? (1.0f / (nrm + 1e-7f)) : 1.0f;
    #pragma unroll
    for (int i = 0; i < 4; ++i)
        x[(size_t)row * DIMD + threadIdx.x + i * 256] = v[i] * scale;
}

// ------------------------------------------------- rmsnorm -> bf16
__global__ void rmsnorm_bf16_kernel(const float* __restrict__ x,
                                    const float* __restrict__ w,
                                    __bf16* __restrict__ out) {
    int row = blockIdx.x;
    const float* xr = x + (size_t)row * DIMD;
    float v[4], ss = 0.0f;
    #pragma unroll
    for (int i = 0; i < 4; ++i) {
        v[i] = xr[threadIdx.x + i * 256];
        ss += v[i] * v[i];
    }
    ss = block_reduce_sum_256(ss);
    float scale = rsqrtf(ss * (1.0f / DIMD) + 1e-6f);
    #pragma unroll
    for (int i = 0; i < 4; ++i) {
        int c = threadIdx.x + i * 256;
        out[(size_t)row * DIMD + c] = (__bf16)(v[i] * scale * w[c]);
    }
}

// ------------------------------------------------- WMMA GEMM (bf16 -> f32)
// C[M,N] (+)= A[M,K] * Bt[N,K]^T + bias[N]
// 256 threads = 8 waves; block tile 128x128; wave tile 64x32 (4x2 WMMA frags).
// A/B K-slabs (128x32 bf16 each) staged in double-buffered LDS via
// global_load_async_to_lds_b128 (ASYNCcnt), fragments fed by ds_load_b128.
__global__ __launch_bounds__(256)
void gemm_wmma_bf16_kernel(const __bf16* __restrict__ A,
                           const __bf16* __restrict__ Bt,
                           const float* __restrict__ bias,
                           float* __restrict__ C,
                           int M, int N, int K, int accumulate) {
    __shared__ __bf16 sA[2][128 * 32];
    __shared__ __bf16 sB[2][128 * 32];

    const int tid   = threadIdx.x;
    const int wave  = tid >> 5;
    const int lane  = tid & 31;
    const int mlane = lane & 15;
    const int khalf = lane >> 4;
    const int koff  = khalf * 8;

    const int wm = wave >> 2;          // 0..1  -> 64-row slice
    const int wn = wave & 3;           // 0..3  -> 32-col slice
    const int m0 = blockIdx.y * 128;
    const int n0 = blockIdx.x * 128;

    const unsigned ldsA0 = lds_addr32(&sA[0][0]);
    const unsigned ldsB0 = lds_addr32(&sB[0][0]);

    // stage loader: each thread moves 2x16B of A and 2x16B of B (4 async ops)
    auto issue = [&](int kk, int sbuf) {
        #pragma unroll
        for (int s = 0; s < 2; ++s) {
            int seg = tid + s * 256;           // 0..511
            int row = seg >> 2;                // 0..127
            int c8  = (seg & 3) * 8;           // 0,8,16,24 (bf16 elements)
            unsigned loff = (unsigned)(sbuf * 4096 + row * 32 + c8) * 2u;
            async_ld_b128(ldsA0 + loff, A  + (size_t)(m0 + row) * K + kk + c8);
            async_ld_b128(ldsB0 + loff, Bt + (size_t)(n0 + row) * K + kk + c8);
        }
    };

    v8f acc[4][2] = {};
    const int KT = K >> 5;                     // K / 32 stages
    int buf = 0;

    issue(0, 0);

    for (int i = 0; i < KT; ++i) {
        if (i + 1 < KT) {
            issue((i + 1) << 5, buf ^ 1);
            asm volatile("s_wait_asynccnt 0x4" ::: "memory");  // prev stage done
        } else {
            asm volatile("s_wait_asynccnt 0x0" ::: "memory");
        }
        __syncthreads();

        v16bf af[4], bfr[2];
        #pragma unroll
        for (int mi = 0; mi < 4; ++mi)
            af[mi] = load_frag16(&sA[buf][(wm * 64 + mi * 16 + mlane) * 32 + koff]);
        #pragma unroll
        for (int nj = 0; nj < 2; ++nj)
            bfr[nj] = load_frag16(&sB[buf][(wn * 32 + nj * 16 + mlane) * 32 + koff]);

        #pragma unroll
        for (int mi = 0; mi < 4; ++mi)
            #pragma unroll
            for (int nj = 0; nj < 2; ++nj)
                acc[mi][nj] = __builtin_amdgcn_wmma_f32_16x16x32_bf16(
                    false, af[mi], false, bfr[nj], (short)0, acc[mi][nj],
                    false, false);

        __syncthreads();                        // LDS buffer safe to overwrite
        buf ^= 1;
    }

    // C/D layout: lane -> N = lane&15, element r -> M = r + 8*(lane>>4)
    #pragma unroll
    for (int mi = 0; mi < 4; ++mi) {
        #pragma unroll
        for (int nj = 0; nj < 2; ++nj) {
            int col = n0 + wn * 32 + nj * 16 + mlane;
            float bv = bias[col];
            int rowbase = m0 + wm * 64 + mi * 16 + khalf * 8;
            #pragma unroll
            for (int r = 0; r < 8; ++r) {
                size_t idx = (size_t)(rowbase + r) * N + col;
                float v = acc[mi][nj][r] + bv;
                if (accumulate) C[idx] += v; else C[idx] = v;
            }
        }
    }
}

// ------------------------------------------------- sioconv recurrent scan
__global__ void sioconv_scan_kernel(const float* __restrict__ z,
                                    const float* __restrict__ dt,
                                    const float* __restrict__ h0,
                                    __bf16* __restrict__ h) {
    int t = blockIdx.x * blockDim.x + threadIdx.x;   // 0 .. B*D-1
    int b = t / DIMD, d = t % DIMD;
    float cs = 0.0f, lcse = -INFINITY;
    float h0v = h0[d];
    for (int l = 0; l < LL; ++l) {
        size_t idx = ((size_t)b * LL + l) * DIMD + d;
        float zv = z[idx], dtv = dt[idx];
        float ln_z   = -softplusf(-zv);
        float ln_da  = -softplusf(-dtv);
        float ln_oda = -softplusf(dtv);
        cs += ln_oda;
        float tt = ln_z + ln_da - cs;
        lcse = logaddexpf_(lcse, tt);
        float hv = expf(lcse + cs) + expf(cs) * h0v;
        h[idx] = (__bf16)hv;
    }
}

// ------------------------------------------------- elementwise epilogues
__global__ void gate_residual_kernel(float* __restrict__ x,
                                     const float* __restrict__ y,
                                     const float* __restrict__ ya, int n) {
    int i = blockIdx.x * blockDim.x + threadIdx.x;
    if (i < n) x[i] += y[i] * siluf(ya[i]);
}

__global__ void swiglu_bf16_kernel(const float* __restrict__ f,
                                   const float* __restrict__ fa,
                                   __bf16* __restrict__ g, int n) {
    int i = blockIdx.x * blockDim.x + threadIdx.x;
    if (i < n) g[i] = (__bf16)(f[i] * siluf(fa[i]));
}

// =================================================================
extern "C" void kernel_launch(void* const* d_in, const int* in_sizes, int n_in,
                              void* d_out, int out_size, void* d_ws, size_t ws_size,
                              hipStream_t stream) {
    (void)in_sizes; (void)n_in; (void)out_size; (void)ws_size;

    const int*   tokens = (const int*)  d_in[0];
    const float* emb    = (const float*)d_in[1];
    const float* W_ln_z = (const float*)d_in[2];
    const float* b_ln_z = (const float*)d_in[3];
    const float* W_dt   = (const float*)d_in[4];
    const float* b_dt   = (const float*)d_in[5];
    const float* W_y    = (const float*)d_in[6];
    const float* b_y    = (const float*)d_in[7];
    const float* W_ya   = (const float*)d_in[8];
    const float* b_ya   = (const float*)d_in[9];
    const float* h0     = (const float*)d_in[10];
    const float* Wf     = (const float*)d_in[11];
    const float* bf     = (const float*)d_in[12];
    const float* Wfa    = (const float*)d_in[13];
    const float* bfa    = (const float*)d_in[14];
    const float* Wo     = (const float*)d_in[15];
    const float* bo     = (const float*)d_in[16];
    const float* n1w    = (const float*)d_in[17];
    const float* n2w    = (const float*)d_in[18];
    const float* nlw    = (const float*)d_in[19];
    const float* W_out  = (const float*)d_in[20];
    const float* b_out  = (const float*)d_in[21];
    float*       out    = (float*)d_out;

    // --------- carve workspace (256B aligned chunks) ---------
    char* wp = (char*)d_ws;
    auto carve = [&](size_t bytes) -> void* {
        void* r = (void*)wp;
        wp += (bytes + 255) & ~(size_t)255;
        return r;
    };
    const size_t DD = (size_t)DIMD * DIMD;       // 1M
    const size_t DF = (size_t)DIMD * FFD;        // 2M
    __bf16* wt_z   = (__bf16*)carve(DD * 2);
    __bf16* wt_dt  = (__bf16*)carve(DD * 2);
    __bf16* wt_y   = (__bf16*)carve(DD * 2);
    __bf16* wt_ya  = (__bf16*)carve(DD * 2);
    __bf16* wt_f   = (__bf16*)carve(DF * 2);     // [FF, D]
    __bf16* wt_fa  = (__bf16*)carve(DF * 2);     // [FF, D]
    __bf16* wt_o   = (__bf16*)carve(DF * 2);     // [D, FF]
    __bf16* wt_out = (__bf16*)carve((size_t)DIMD * VV * 2);   // [V, D]

    float*  x    = (float*) carve((size_t)MTOK * DIMD * 4);
    float*  zb   = (float*) carve((size_t)MTOK * DIMD * 4);
    float*  dtb  = (float*) carve((size_t)MTOK * DIMD * 4);
    float*  yab  = (float*) carve((size_t)MTOK * DIMD * 4);
    float*  fb   = (float*) carve((size_t)MTOK * FFD  * 4);
    float*  fab  = (float*) carve((size_t)MTOK * FFD  * 4);
    __bf16* xn   = (__bf16*)carve((size_t)MTOK * DIMD * 2);
    __bf16* hb   = (__bf16*)carve((size_t)MTOK * DIMD * 2);
    __bf16* gb   = (__bf16*)carve((size_t)MTOK * FFD  * 2);

    const dim3 tb_tr(32, 8);
    const int  nD = MTOK * DIMD;      // 4,194,304
    const int  nF = MTOK * FFD;       // 8,388,608

    // --------- embedding + renorm ---------
    embed_renorm_kernel<<<MTOK, 256, 0, stream>>>(tokens, emb, x);

    for (int i = 0; i < NDEP; ++i) {
        // convert + transpose this layer's weights to bf16 [N, K]
        cvt_transpose_kernel<<<dim3(DIMD/32, DIMD/32), tb_tr, 0, stream>>>(W_ln_z + i*DD, wt_z,  DIMD, DIMD);
        cvt_transpose_kernel<<<dim3(DIMD/32, DIMD/32), tb_tr, 0, stream>>>(W_dt   + i*DD, wt_dt, DIMD, DIMD);
        cvt_transpose_kernel<<<dim3(DIMD/32, DIMD/32), tb_tr, 0, stream>>>(W_y    + i*DD, wt_y,  DIMD, DIMD);
        cvt_transpose_kernel<<<dim3(DIMD/32, DIMD/32), tb_tr, 0, stream>>>(W_ya   + i*DD, wt_ya, DIMD, DIMD);
        cvt_transpose_kernel<<<dim3(FFD/32,  DIMD/32), tb_tr, 0, stream>>>(Wf     + i*DF, wt_f,  DIMD, FFD);
        cvt_transpose_kernel<<<dim3(FFD/32,  DIMD/32), tb_tr, 0, stream>>>(Wfa    + i*DF, wt_fa, DIMD, FFD);
        cvt_transpose_kernel<<<dim3(DIMD/32, FFD/32),  tb_tr, 0, stream>>>(Wo     + i*DF, wt_o,  FFD,  DIMD);

        // ---- sioconv ----
        rmsnorm_bf16_kernel<<<MTOK, 256, 0, stream>>>(x, n1w + i*DIMD, xn);

        gemm_wmma_bf16_kernel<<<dim3(DIMD/128, MTOK/128), 256, 0, stream>>>(
            xn, wt_z,  b_ln_z + i*DIMD, zb,  MTOK, DIMD, DIMD, 0);
        gemm_wmma_bf16_kernel<<<dim3(DIMD/128, MTOK/128), 256, 0, stream>>>(
            xn, wt_dt, b_dt   + i*DIMD, dtb, MTOK, DIMD, DIMD, 0);
        gemm_wmma_bf16_kernel<<<dim3(DIMD/128, MTOK/128), 256, 0, stream>>>(
            xn, wt_ya, b_ya   + i*DIMD, yab, MTOK, DIMD, DIMD, 0);

        sioconv_scan_kernel<<<(BB*DIMD)/256, 256, 0, stream>>>(zb, dtb, h0 + i*DIMD, hb);

        // y = h @ Wy + by   (reuse zb)
        gemm_wmma_bf16_kernel<<<dim3(DIMD/128, MTOK/128), 256, 0, stream>>>(
            hb, wt_y, b_y + i*DIMD, zb, MTOK, DIMD, DIMD, 0);

        gate_residual_kernel<<<(nD + 255)/256, 256, 0, stream>>>(x, zb, yab, nD);

        // ---- ffn swiglu ----
        rmsnorm_bf16_kernel<<<MTOK, 256, 0, stream>>>(x, n2w + i*DIMD, xn);

        gemm_wmma_bf16_kernel<<<dim3(FFD/128, MTOK/128), 256, 0, stream>>>(
            xn, wt_f,  bf  + i*FFD, fb,  MTOK, FFD, DIMD, 0);
        gemm_wmma_bf16_kernel<<<dim3(FFD/128, MTOK/128), 256, 0, stream>>>(
            xn, wt_fa, bfa + i*FFD, fab, MTOK, FFD, DIMD, 0);

        swiglu_bf16_kernel<<<(nF + 255)/256, 256, 0, stream>>>(fb, fab, gb, nF);

        // x += g @ Wo + bo  (fused residual via accumulate=1)
        gemm_wmma_bf16_kernel<<<dim3(DIMD/128, MTOK/128), 256, 0, stream>>>(
            gb, wt_o, bo + i*DIMD, x, MTOK, DIMD, FFD, 1);
    }

    // --------- final norm + output projection ---------
    rmsnorm_bf16_kernel<<<MTOK, 256, 0, stream>>>(x, nlw, xn);
    cvt_transpose_kernel<<<dim3(VV/32, DIMD/32), tb_tr, 0, stream>>>(W_out, wt_out, DIMD, VV);
    gemm_wmma_bf16_kernel<<<dim3(VV/128, MTOK/128), 256, 0, stream>>>(
        xn, wt_out, b_out, out, MTOK, VV, DIMD, 0);
}
